// GATModel_82721070121591
// MI455X (gfx1250) — matrix operand order repaired and checked by reading, not compiled, not run on hardware
//
#include <hip/hip_runtime.h>
#include <hip/hip_bf16.h>

#define N_NODES 50000
#define E_EDGES 1600000
#define E_TOT   (E_EDGES + N_NODES)   // with self-loops
#define F_IN    512
#define H1      8
#define C1      8
#define D1      64                    // H1*C1
#define OUTC    128
#define NEG_SLOPE 0.2f
#define ROW_TILES (N_NODES / 16)      // 3125

typedef __attribute__((ext_vector_type(16))) _Float16 v16h;
typedef __attribute__((ext_vector_type(8)))  _Float16 v8h;
typedef __attribute__((ext_vector_type(8)))  float    v8f;

union AF { v16h v; v8h h[2]; _Float16 e[16]; };
union CF { v8f  v; float e[8]; };

// order-preserving float <-> uint mapping (for atomic segment-max)
static __device__ __forceinline__ unsigned fmap(float f) {
    unsigned u = __float_as_uint(f);
    return (u & 0x80000000u) ? ~u : (u | 0x80000000u);
}
static __device__ __forceinline__ float funmap(unsigned m) {
    return __uint_as_float((m & 0x80000000u) ? (m & 0x7fffffffu) : ~m);
}
static __device__ __forceinline__ void edge_sd(const int* __restrict__ ei, int e, int& s, int& d) {
    if (e < E_EDGES) { s = ei[e]; d = ei[E_EDGES + e]; }
    else             { s = e - E_EDGES; d = s; }
}

// ---------------- weight prep: transpose + f32->f16 ----------------
__global__ void k_prep_weights(const float* __restrict__ W1, const float* __restrict__ W2,
                               _Float16* __restrict__ w1t, _Float16* __restrict__ w2t) {
    int i = blockIdx.x * blockDim.x + threadIdx.x;
    if (i < D1 * F_IN) {                 // w1t[n][k] = W1[k][n]
        int n = i / F_IN, k = i % F_IN;
        w1t[i] = (_Float16)W1[k * D1 + n];
    } else {
        int j = i - D1 * F_IN;
        if (j < OUTC * D1) {             // w2t[n][k] = W2[k][n]
            int n = j / D1, k = j % D1;
            w2t[j] = (_Float16)W2[k * OUTC + n];
        }
    }
}

// ---------------- GEMM1: h1[N,64] = x[N,512] * W1 (f16 WMMA, f32 acc) ----------------
// one wave owns one 16-row tile; A fragment reused across all 4 column tiles
__global__ void __launch_bounds__(128) k_gemm1(const float* __restrict__ x,
                                               const _Float16* __restrict__ w1t,
                                               float* __restrict__ h1) {
    const int lane = threadIdx.x & 31;
    const int wave = threadIdx.x >> 5;
    const int rowTile = blockIdx.x * 4 + wave;
    if (rowTile >= ROW_TILES) return;              // wave-uniform exit
    const int mn   = lane & 15;
    const int half = lane >> 4;
    const int rowBase = rowTile * 16;
    const float* xrow = x + (size_t)(rowBase + mn) * F_IN;

    CF acc[4];
    #pragma unroll
    for (int ct = 0; ct < 4; ++ct) acc[ct].v = (v8f){};

    for (int kt = 0; kt < F_IN / 32; ++kt) {
        const int kb = kt * 32;
        // A fragment: one contiguous 16-elem run per lane, cvt f32->f16 once
        const float4 f0 = *(const float4*)(xrow + kb + half * 16);
        const float4 f1 = *(const float4*)(xrow + kb + half * 16 + 4);
        const float4 f2 = *(const float4*)(xrow + kb + half * 16 + 8);
        const float4 f3 = *(const float4*)(xrow + kb + half * 16 + 12);
        AF a;
        a.e[0]=(_Float16)f0.x;  a.e[1]=(_Float16)f0.y;  a.e[2]=(_Float16)f0.z;  a.e[3]=(_Float16)f0.w;
        a.e[4]=(_Float16)f1.x;  a.e[5]=(_Float16)f1.y;  a.e[6]=(_Float16)f1.z;  a.e[7]=(_Float16)f1.w;
        a.e[8]=(_Float16)f2.x;  a.e[9]=(_Float16)f2.y;  a.e[10]=(_Float16)f2.z; a.e[11]=(_Float16)f2.w;
        a.e[12]=(_Float16)f3.x; a.e[13]=(_Float16)f3.y; a.e[14]=(_Float16)f3.z; a.e[15]=(_Float16)f3.w;
        #pragma unroll
        for (int ct = 0; ct < 4; ++ct) {
            v16h b = *(const v16h*)(w1t + (size_t)(ct * 16 + mn) * F_IN + kb + half * 16);
            acc[ct].v = __builtin_amdgcn_wmma_f32_16x16x32_f16(false, a.v, false, b,
                                                               (short)0, acc[ct].v, false, false);
        }
    }
    #pragma unroll
    for (int ct = 0; ct < 4; ++ct) {
        float* orow = h1 + (size_t)(rowBase + half * 8) * D1 + ct * 16 + mn;
        #pragma unroll
        for (int r = 0; r < 8; ++r) orow[(size_t)r * D1] = acc[ct].e[r];
    }
}

// ---------------- attention coefficients, layer 1 ----------------
__global__ void k_alpha1(const float* __restrict__ h1, const float* __restrict__ a_src,
                         const float* __restrict__ a_dst,
                         float* __restrict__ as1, float* __restrict__ ad1) {
    int i = blockIdx.x * blockDim.x + threadIdx.x;   // node*8 + head
    if (i >= N_NODES * H1) return;
    int hh = i & 7, n = i >> 3;
    const float* hp = h1 + (size_t)n * D1 + hh * C1;
    float s = 0.f, d = 0.f;
    #pragma unroll
    for (int c = 0; c < C1; ++c) {
        s += hp[c] * a_src[hh * C1 + c];
        d += hp[c] * a_dst[hh * C1 + c];
    }
    as1[i] = s; ad1[i] = d;
}

__global__ void k_edge_max1(const int* __restrict__ ei, const float* __restrict__ as1,
                            const float* __restrict__ ad1, unsigned* __restrict__ smax) {
    long long i = (long long)blockIdx.x * blockDim.x + threadIdx.x;
    if (i >= (long long)E_TOT * H1) return;
    int e = (int)(i >> 3), hh = (int)(i & 7), s, d;
    edge_sd(ei, e, s, d);
    float ev = as1[s * H1 + hh] + ad1[d * H1 + hh];
    ev = ev > 0.f ? ev : NEG_SLOPE * ev;
    atomicMax(&smax[d * H1 + hh], fmap(ev));
}

__global__ void k_edge_denom1(const int* __restrict__ ei, const float* __restrict__ as1,
                              const float* __restrict__ ad1, const unsigned* __restrict__ smax,
                              float* __restrict__ denom) {
    long long i = (long long)blockIdx.x * blockDim.x + threadIdx.x;
    if (i >= (long long)E_TOT * H1) return;
    int e = (int)(i >> 3), hh = (int)(i & 7), s, d;
    edge_sd(ei, e, s, d);
    float ev = as1[s * H1 + hh] + ad1[d * H1 + hh];
    ev = ev > 0.f ? ev : NEG_SLOPE * ev;
    atomicAdd(&denom[d * H1 + hh], expf(ev - funmap(smax[d * H1 + hh])));
}

__global__ void k_msg1(const int* __restrict__ ei, const float* __restrict__ as1,
                       const float* __restrict__ ad1, const unsigned* __restrict__ smax,
                       const float* __restrict__ denom, const float* __restrict__ h1,
                       float* __restrict__ agg) {
    long long i = (long long)blockIdx.x * blockDim.x + threadIdx.x;
    if (i >= (long long)E_TOT * H1) return;
    int e = (int)(i >> 3), hh = (int)(i & 7), s, d;
    edge_sd(ei, e, s, d);
    float ev = as1[s * H1 + hh] + ad1[d * H1 + hh];
    ev = ev > 0.f ? ev : NEG_SLOPE * ev;
    float coef = expf(ev - funmap(smax[d * H1 + hh])) / denom[d * H1 + hh];
    const float* hs = h1  + (size_t)s * D1 + hh * C1;
    float*       od = agg + (size_t)d * D1 + hh * C1;
    #pragma unroll
    for (int c = 0; c < C1; ++c) atomicAdd(&od[c], hs[c] * coef);
}

// bias + ELU + cvt to f16 feature for layer 2
__global__ void k_fin1(const float* __restrict__ agg, const float* __restrict__ b1,
                       _Float16* __restrict__ f2h) {
    int i = blockIdx.x * blockDim.x + threadIdx.x;
    if (i >= N_NODES * D1) return;
    float v = agg[i] + b1[i & (D1 - 1)];
    v = v > 0.f ? v : (expf(v) - 1.f);
    f2h[i] = (_Float16)v;
}

// ---------------- GEMM2: h2[N,128] = f2[N,64] * W2 (f16 WMMA) ----------------
// one wave owns one 16-row tile; A reused across all 8 column tiles
__global__ void __launch_bounds__(128) k_gemm2(const _Float16* __restrict__ f2h,
                                               const _Float16* __restrict__ w2t,
                                               float* __restrict__ h2) {
    const int lane = threadIdx.x & 31;
    const int wave = threadIdx.x >> 5;
    const int rowTile = blockIdx.x * 4 + wave;
    if (rowTile >= ROW_TILES) return;              // wave-uniform exit
    const int mn   = lane & 15;
    const int half = lane >> 4;
    const int rowBase = rowTile * 16;
    const _Float16* arow = f2h + (size_t)(rowBase + mn) * D1;

    CF acc[8];
    #pragma unroll
    for (int ct = 0; ct < 8; ++ct) acc[ct].v = (v8f){};

    #pragma unroll
    for (int kt = 0; kt < D1 / 32; ++kt) {
        const int kb = kt * 32;
        AF a;
        a.h[0] = *(const v8h*)(arow + kb + half * 16);
        a.h[1] = *(const v8h*)(arow + kb + half * 16 + 8);
        #pragma unroll
        for (int ct = 0; ct < 8; ++ct) {
            v16h b = *(const v16h*)(w2t + (size_t)(ct * 16 + mn) * D1 + kb + half * 16);
            acc[ct].v = __builtin_amdgcn_wmma_f32_16x16x32_f16(false, a.v, false, b,
                                                               (short)0, acc[ct].v, false, false);
        }
    }
    #pragma unroll
    for (int ct = 0; ct < 8; ++ct) {
        float* orow = h2 + (size_t)(rowBase + half * 8) * OUTC + ct * 16 + mn;
        #pragma unroll
        for (int r = 0; r < 8; ++r) orow[(size_t)r * OUTC] = acc[ct].e[r];
    }
}

__global__ void k_alpha2(const float* __restrict__ h2, const float* __restrict__ a_src,
                         const float* __restrict__ a_dst,
                         float* __restrict__ as2, float* __restrict__ ad2) {
    int n = blockIdx.x * blockDim.x + threadIdx.x;
    if (n >= N_NODES) return;
    const float* hp = h2 + (size_t)n * OUTC;
    float s = 0.f, d = 0.f;
    for (int c = 0; c < OUTC; ++c) { s += hp[c] * a_src[c]; d += hp[c] * a_dst[c]; }
    as2[n] = s; ad2[n] = d;
}

__global__ void k_edge_max2(const int* __restrict__ ei, const float* __restrict__ as2,
                            const float* __restrict__ ad2, unsigned* __restrict__ smax) {
    int e = blockIdx.x * blockDim.x + threadIdx.x;
    if (e >= E_TOT) return;
    int s, d; edge_sd(ei, e, s, d);
    float ev = as2[s] + ad2[d];
    ev = ev > 0.f ? ev : NEG_SLOPE * ev;
    atomicMax(&smax[d], fmap(ev));
}

__global__ void k_edge_denom2(const int* __restrict__ ei, const float* __restrict__ as2,
                              const float* __restrict__ ad2, const unsigned* __restrict__ smax,
                              float* __restrict__ denom) {
    int e = blockIdx.x * blockDim.x + threadIdx.x;
    if (e >= E_TOT) return;
    int s, d; edge_sd(ei, e, s, d);
    float ev = as2[s] + ad2[d];
    ev = ev > 0.f ? ev : NEG_SLOPE * ev;
    atomicAdd(&denom[d], expf(ev - funmap(smax[d])));
}

__global__ void k_msg2(const int* __restrict__ ei, const float* __restrict__ as2,
                       const float* __restrict__ ad2, const unsigned* __restrict__ smax,
                       const float* __restrict__ denom, const float* __restrict__ h2,
                       float* __restrict__ out) {
    long long i = (long long)blockIdx.x * blockDim.x + threadIdx.x;
    if (i >= (long long)E_TOT * 8) return;
    int e = (int)(i >> 3), chunk = (int)(i & 7), s, d;
    edge_sd(ei, e, s, d);
    float ev = as2[s] + ad2[d];
    ev = ev > 0.f ? ev : NEG_SLOPE * ev;
    float coef = expf(ev - funmap(smax[d])) / denom[d];
    const float* hs = h2  + (size_t)s * OUTC + chunk * 16;
    float*       od = out + (size_t)d * OUTC + chunk * 16;
    #pragma unroll
    for (int c = 0; c < 16; ++c) atomicAdd(&od[c], hs[c] * coef);
}

__global__ void k_fin2(float* __restrict__ out, const float* __restrict__ b2) {
    int i = blockIdx.x * blockDim.x + threadIdx.x;
    if (i >= N_NODES * OUTC) return;
    out[i] += b2[i & (OUTC - 1)];
}

// ---------------- launch ----------------
extern "C" void kernel_launch(void* const* d_in, const int* in_sizes, int n_in,
                              void* d_out, int out_size, void* d_ws, size_t ws_size,
                              hipStream_t stream) {
    (void)in_sizes; (void)n_in; (void)out_size; (void)ws_size;
    const float* x     = (const float*)d_in[0];
    const int*   ei    = (const int*)  d_in[1];
    const float* W1    = (const float*)d_in[2];
    const float* asr1  = (const float*)d_in[3];
    const float* ads1  = (const float*)d_in[4];
    const float* b1    = (const float*)d_in[5];
    const float* W2    = (const float*)d_in[6];
    const float* asr2  = (const float*)d_in[7];
    const float* ads2  = (const float*)d_in[8];
    const float* b2    = (const float*)d_in[9];
    float* out = (float*)d_out;

    char* ws = (char*)d_ws;
    size_t off = 0;
    auto alloc = [&](size_t bytes) -> char* {
        char* p = ws + off;
        off = (off + bytes + 255) & ~(size_t)255;
        return p;
    };
    _Float16* w1t   = (_Float16*)alloc((size_t)D1 * F_IN * 2);
    _Float16* w2t   = (_Float16*)alloc((size_t)OUTC * D1 * 2);
    float*    h1    = (float*)   alloc((size_t)N_NODES * D1 * 4);
    float*    as1   = (float*)   alloc((size_t)N_NODES * H1 * 4);
    float*    ad1   = (float*)   alloc((size_t)N_NODES * H1 * 4);
    unsigned* smax1 = (unsigned*)alloc((size_t)N_NODES * H1 * 4);
    float*    den1  = (float*)   alloc((size_t)N_NODES * H1 * 4);
    float*    agg1  = (float*)   alloc((size_t)N_NODES * D1 * 4);
    _Float16* f2h   = (_Float16*)alloc((size_t)N_NODES * D1 * 2);
    float*    h2    = (float*)   alloc((size_t)N_NODES * OUTC * 4);
    float*    as2   = (float*)   alloc((size_t)N_NODES * 4);
    float*    ad2   = (float*)   alloc((size_t)N_NODES * 4);
    unsigned* smax2 = (unsigned*)alloc((size_t)N_NODES * 4);
    float*    den2  = (float*)   alloc((size_t)N_NODES * 4);

    hipMemsetAsync(smax1, 0, (size_t)N_NODES * H1 * 4, stream);
    hipMemsetAsync(den1,  0, (size_t)N_NODES * H1 * 4, stream);
    hipMemsetAsync(agg1,  0, (size_t)N_NODES * D1 * 4, stream);
    hipMemsetAsync(smax2, 0, (size_t)N_NODES * 4, stream);
    hipMemsetAsync(den2,  0, (size_t)N_NODES * 4, stream);
    hipMemsetAsync(out,   0, (size_t)N_NODES * OUTC * 4, stream);

    const int T = 256;
    const long long eh = (long long)E_TOT * 8;
    const int ehBlocks = (int)((eh + T - 1) / T);
    const int gemmBlocks = (ROW_TILES + 3) / 4;   // 4 waves/block, 1 row tile/wave

    k_prep_weights<<<(D1 * F_IN + OUTC * D1 + T - 1) / T, T, 0, stream>>>(W1, W2, w1t, w2t);
    k_gemm1<<<gemmBlocks, 128, 0, stream>>>(x, w1t, h1);
    k_alpha1<<<(N_NODES * H1 + T - 1) / T, T, 0, stream>>>(h1, asr1, ads1, as1, ad1);
    k_edge_max1<<<ehBlocks, T, 0, stream>>>(ei, as1, ad1, smax1);
    k_edge_denom1<<<ehBlocks, T, 0, stream>>>(ei, as1, ad1, smax1, den1);
    k_msg1<<<ehBlocks, T, 0, stream>>>(ei, as1, ad1, smax1, den1, h1, agg1);
    k_fin1<<<(N_NODES * D1 + T - 1) / T, T, 0, stream>>>(agg1, b1, f2h);
    k_gemm2<<<gemmBlocks, 128, 0, stream>>>(f2h, w2t, h2);
    k_alpha2<<<(N_NODES + T - 1) / T, T, 0, stream>>>(h2, asr2, ads2, as2, ad2);
    k_edge_max2<<<(E_TOT + T - 1) / T, T, 0, stream>>>(ei, as2, ad2, smax2);
    k_edge_denom2<<<(E_TOT + T - 1) / T, T, 0, stream>>>(ei, as2, ad2, smax2, den2);
    k_msg2<<<ehBlocks, T, 0, stream>>>(ei, as2, ad2, smax2, den2, h2, out);
    k_fin2<<<(N_NODES * OUTC + T - 1) / T, T, 0, stream>>>(out, b2);
}
